// MoE_10282151707685
// MI455X (gfx1250) — compile-verified, matrix operand order checked by reading, not run on hardware
//
#include <hip/hip_runtime.h>

// ---------------------------------------------------------------------------
// MoE top-2 routed FFN for MI455X (gfx1250, wave32, WMMA bf16 16x16x32)
//   - route first, compute only top-2 experts (58 GFLOP vs 232 dense)
//   - pre-convert/transpose weights to bf16 [N][K] once (WMMA-native frags)
//   - 64-token M tiles; 8 wave32 per block as 2 M-halves x 4 N-subtiles,
//     each wave holds 2 M-tiles so every B fragment feeds 2 WMMAs
// ---------------------------------------------------------------------------

typedef __attribute__((ext_vector_type(16))) __bf16 v16bf;
typedef __attribute__((ext_vector_type(8)))  float  v8f;

#define D_IN        768
#define H_DIM       1536
#define NEXP        8
#define NTOK        4096          // B*T = 2*2048
#define MTILE       64            // tokens per block
#define TILES_PER_E 64            // ceil(4096/64)
#define MAX_ROWS    8704          // 2*NTOK + NEXP*63 padding, rounded up
#define XA_STRIDE   784           // 768 + 16 pad: 1568B row stride (32B aligned)

// ---- workspace layout (bytes) ---------------------------------------------
#define WS_COUNT 0                                       // int[NEXP]
#define WS_OFFS  64                                      // int[NEXP]
#define WS_LIST  256                                     // int[NEXP*NTOK]
#define WS_SEL_E (WS_LIST + NEXP*NTOK*4)                 // int[NTOK*2]
#define WS_SEL_P (WS_SEL_E + NTOK*2*4)                   // int[NTOK*2]
#define WS_SEL_W (WS_SEL_P + NTOK*2*4)                   // float[NTOK*2]
#define WS_WG_T  (WS_SEL_W + NTOK*2*4)                   // bf16[E][H][D]
#define WS_WU_T  (WS_WG_T + (size_t)NEXP*D_IN*H_DIM*2)   // bf16[E][H][D]
#define WS_WD_T  (WS_WU_T + (size_t)NEXP*D_IN*H_DIM*2)   // bf16[E][D][H]
#define WS_H     (WS_WD_T + (size_t)NEXP*D_IN*H_DIM*2)   // bf16[MAX_ROWS][H]
#define WS_Y     (WS_H + (size_t)MAX_ROWS*H_DIM*2)       // f32 [MAX_ROWS][D]

static __device__ __forceinline__ v8f wmma_bf16(v16bf a, v16bf b, v8f c) {
    return __builtin_amdgcn_wmma_f32_16x16x32_bf16(
        false, a, false, b, (short)0, c, false, false);
}

// ---------------------------------------------------------------------------
// Kernel 0: zero the per-expert counters
// ---------------------------------------------------------------------------
__global__ void moe_zero(int* __restrict__ counts) {
    if (threadIdx.x < NEXP) counts[threadIdx.x] = 0;
}

// ---------------------------------------------------------------------------
// Kernel T: per-expert tiled transpose + fp32->bf16:  in[e][R][C] -> out[e][C][R]
// ---------------------------------------------------------------------------
__global__ __launch_bounds__(256)
void transpose_cvt(const float* __restrict__ in, __bf16* __restrict__ out,
                   int rows, int cols) {
    int br = rows >> 5, bc = cols >> 5;
    int id  = blockIdx.x;
    int e   = id / (br * bc);
    int rem = id - e * (br * bc);
    int tr  = rem / bc;
    int tc  = rem - tr * bc;
    const float* src = in  + (size_t)e * rows * cols;
    __bf16*      dst = out + (size_t)e * rows * cols;

    __shared__ __bf16 tilebuf[32][33];
    int t = threadIdx.x;
    int c = t & 31, r0 = t >> 5;
#pragma unroll
    for (int k = 0; k < 4; ++k) {
        int r = r0 + k * 8;
        tilebuf[r][c] = (__bf16)src[(size_t)(tr * 32 + r) * cols + tc * 32 + c];
    }
    __syncthreads();
#pragma unroll
    for (int k = 0; k < 4; ++k) {
        int oc  = t & 31;       // original row index
        int orr = r0 + k * 8;   // original col index
        dst[(size_t)(tc * 32 + orr) * rows + tr * 32 + oc] = tilebuf[oc][orr];
    }
}

// ---------------------------------------------------------------------------
// Kernel 1: gating — logits, softmax, top-2, renormalized weights, lists
// ---------------------------------------------------------------------------
__global__ __launch_bounds__(256)
void moe_gate(const float* __restrict__ x, const float* __restrict__ gw,
              int* __restrict__ counts, int* __restrict__ list,
              int* __restrict__ sel_e, int* __restrict__ sel_p,
              float* __restrict__ sel_w) {
    int t = blockIdx.x * 256 + threadIdx.x;
    if (t >= NTOK) return;
    const float* xr = x + (size_t)t * D_IN;

    float l[NEXP];
#pragma unroll
    for (int e = 0; e < NEXP; ++e) l[e] = 0.f;
    for (int d = 0; d < D_IN; ++d) {
        float xv = xr[d];
#pragma unroll
        for (int e = 0; e < NEXP; ++e) l[e] += xv * gw[e * D_IN + d];
    }
    int i0 = 0;
#pragma unroll
    for (int e = 1; e < NEXP; ++e) if (l[e] > l[i0]) i0 = e;
    int i1 = (i0 == 0) ? 1 : 0;
#pragma unroll
    for (int e = 0; e < NEXP; ++e) if (e != i0 && l[e] > l[i1]) i1 = e;

    float mx = l[i0];
    float Z = 0.f;
#pragma unroll
    for (int e = 0; e < NEXP; ++e) Z += __expf(l[e] - mx);
    float p0 = __expf(l[i0] - mx) / Z;
    float p1 = __expf(l[i1] - mx) / Z;
    float inv = 1.f / (p0 + p1 + 1e-8f);

    int pos0 = atomicAdd(&counts[i0], 1);
    int pos1 = atomicAdd(&counts[i1], 1);
    list[i0 * NTOK + pos0] = t;
    list[i1 * NTOK + pos1] = t;

    sel_e[2 * t + 0] = i0;  sel_p[2 * t + 0] = pos0;  sel_w[2 * t + 0] = p0 * inv;
    sel_e[2 * t + 1] = i1;  sel_p[2 * t + 1] = pos1;  sel_w[2 * t + 1] = p1 * inv;
}

// ---------------------------------------------------------------------------
// Kernel 1.5: 64-padded exclusive prefix of counts -> row offsets per expert
// ---------------------------------------------------------------------------
__global__ void moe_offsets(const int* __restrict__ counts, int* __restrict__ offs) {
    if (threadIdx.x == 0 && blockIdx.x == 0) {
        int run = 0;
        for (int e = 0; e < NEXP; ++e) {
            offs[e] = run;
            run += ((counts[e] + MTILE - 1) / MTILE) * MTILE;
        }
    }
}

// ---------------------------------------------------------------------------
// Kernel 2: gate & up projections + SiLU per (expert, 64-token tile)
// ---------------------------------------------------------------------------
__global__ __launch_bounds__(256)
void moe_gateup(const float* __restrict__ x,
                const __bf16* __restrict__ wg_t,   // [E][H][D] bf16
                const __bf16* __restrict__ wu_t,   // [E][H][D] bf16
                const int* __restrict__ counts, const int* __restrict__ offs,
                const int* __restrict__ list, __bf16* __restrict__ hbuf) {
    int e    = blockIdx.x / TILES_PER_E;
    int tile = blockIdx.x % TILES_PER_E;
    int cnt  = counts[e];
    int row0 = tile * MTILE;
    if (row0 >= cnt) return;

    __shared__ int    toks[MTILE];
    __shared__ __bf16 xa[MTILE * XA_STRIDE];

    int tid = threadIdx.x;
    if (tid < MTILE) {
        int r = row0 + tid;
        toks[tid] = (r < cnt) ? list[e * NTOK + r] : -1;
    }
    __syncthreads();

    for (int idx = tid; idx < MTILE * D_IN; idx += 256) {
        int m = idx / D_IN;
        int d = idx - m * D_IN;
        int tk = toks[m];
        float v = (tk >= 0) ? x[(size_t)tk * D_IN + d] : 0.f;
        xa[m * XA_STRIDE + d] = (__bf16)v;
    }
    __syncthreads();

    const __bf16* gb = wg_t + (size_t)e * D_IN * H_DIM;
    const __bf16* ub = wu_t + (size_t)e * D_IN * H_DIM;

    int wave = tid >> 5, lane = tid & 31;
    int mh   = wave & 1;                     // M half: rows mh*32 .. mh*32+31
    int nt   = wave >> 1;                    // N subtile 0..3 (16 cols each)
    int lrow = lane & 15;                    // M (A) / N (B, C/D)
    int ksel = (lane >> 4) << 4;             // K half select
    int mofs = (lane >> 4) << 3;             // C/D row offset
    int slotrow = offs[e] + row0 + mh * 32;

    const __bf16* arow0 = &xa[(mh * 32 + lrow) * XA_STRIDE + ksel];
    const __bf16* arow1 = arow0 + 16 * XA_STRIDE;

    for (int p = 0; p < H_DIM / 64; ++p) {             // 24 N passes
        int col = p * 64 + nt * 16 + lrow;
        const __bf16* gp = gb + (size_t)col * D_IN + ksel;
        const __bf16* up = ub + (size_t)col * D_IN + ksel;
        // prefetch next pass's weight rows while this pass computes
        __builtin_prefetch(gp + (size_t)64 * D_IN, 0, 1);
        __builtin_prefetch(up + (size_t)64 * D_IN, 0, 1);
        v8f acc_a0 = {}, acc_u0 = {};
        v8f acc_a1 = {}, acc_u1 = {};
#pragma unroll 2
        for (int kk = 0; kk < D_IN; kk += 32) {        // 24 K steps
            v16bf av0 = *(const v16bf*)(arow0 + kk);
            v16bf av1 = *(const v16bf*)(arow1 + kk);
            v16bf bg  = *(const v16bf*)(gp + kk);
            v16bf bu  = *(const v16bf*)(up + kk);
            acc_a0 = wmma_bf16(av0, bg, acc_a0);
            acc_u0 = wmma_bf16(av0, bu, acc_u0);
            acc_a1 = wmma_bf16(av1, bg, acc_a1);
            acc_u1 = wmma_bf16(av1, bu, acc_u1);
        }
#pragma unroll
        for (int r = 0; r < 8; ++r) {
            float a0 = acc_a0[r], u0 = acc_u0[r];
            float h0 = a0 * __builtin_amdgcn_rcpf(1.f + __expf(-a0)) * u0;
            hbuf[(size_t)(slotrow + r + mofs) * H_DIM + col] = (__bf16)h0;
            float a1 = acc_a1[r], u1 = acc_u1[r];
            float h1 = a1 * __builtin_amdgcn_rcpf(1.f + __expf(-a1)) * u1;
            hbuf[(size_t)(slotrow + 16 + r + mofs) * H_DIM + col] = (__bf16)h1;
        }
    }
}

// ---------------------------------------------------------------------------
// Kernel 3: down projection per (expert, 64-token tile):  y = h @ Wd[e]
// ---------------------------------------------------------------------------
__global__ __launch_bounds__(256)
void moe_down(const __bf16* __restrict__ hbuf,
              const __bf16* __restrict__ wd_t,   // [E][D][H] bf16
              const int* __restrict__ counts, const int* __restrict__ offs,
              float* __restrict__ ybuf) {
    int e    = blockIdx.x / TILES_PER_E;
    int tile = blockIdx.x % TILES_PER_E;
    int cnt  = counts[e];
    int row0 = tile * MTILE;
    if (row0 >= cnt) return;

    int slotbase = offs[e] + row0;
    const __bf16* db = wd_t + (size_t)e * H_DIM * D_IN;

    int tid  = threadIdx.x;
    int wave = tid >> 5, lane = tid & 31;
    int mh   = wave & 1;
    int nt   = wave >> 1;
    int lrow = lane & 15;
    int ksel = (lane >> 4) << 4;
    int mofs = (lane >> 4) << 3;

    const __bf16* ap0 =
        hbuf + (size_t)(slotbase + mh * 32 + lrow) * H_DIM + ksel;
    const __bf16* ap1 = ap0 + (size_t)16 * H_DIM;

    for (int p = 0; p < D_IN / 64; ++p) {              // 12 N passes
        int col = p * 64 + nt * 16 + lrow;
        const __bf16* bp = db + (size_t)col * H_DIM + ksel;
        __builtin_prefetch(bp + (size_t)64 * H_DIM, 0, 1);
        v8f acc0 = {};
        v8f acc1 = {};
#pragma unroll 2
        for (int kk = 0; kk < H_DIM; kk += 32) {       // 48 K steps
            v16bf av0 = *(const v16bf*)(ap0 + kk);
            v16bf av1 = *(const v16bf*)(ap1 + kk);
            v16bf bv  = *(const v16bf*)(bp + kk);
            acc0 = wmma_bf16(av0, bv, acc0);
            acc1 = wmma_bf16(av1, bv, acc1);
        }
#pragma unroll
        for (int r = 0; r < 8; ++r) {
            ybuf[(size_t)(slotbase + mh * 32 + r + mofs) * D_IN + col] = acc0[r];
            ybuf[(size_t)(slotbase + mh * 32 + 16 + r + mofs) * D_IN + col] = acc1[r];
        }
    }
}

// ---------------------------------------------------------------------------
// Kernel 4: weighted combine of each token's two expert outputs (deterministic)
// ---------------------------------------------------------------------------
__global__ __launch_bounds__(256)
void moe_combine(const float* __restrict__ ybuf, const int* __restrict__ offs,
                 const int* __restrict__ sel_e, const int* __restrict__ sel_p,
                 const float* __restrict__ sel_w, float* __restrict__ out) {
    int i = blockIdx.x * 256 + threadIdx.x;
    if (i >= NTOK * D_IN) return;
    int t = i / D_IN;
    int d = i - t * D_IN;
    int e0 = sel_e[2 * t + 0], e1 = sel_e[2 * t + 1];
    int s0 = offs[e0] + sel_p[2 * t + 0];
    int s1 = offs[e1] + sel_p[2 * t + 1];
    out[i] = sel_w[2 * t + 0] * ybuf[(size_t)s0 * D_IN + d] +
             sel_w[2 * t + 1] * ybuf[(size_t)s1 * D_IN + d];
}

// ---------------------------------------------------------------------------
extern "C" void kernel_launch(void* const* d_in, const int* in_sizes, int n_in,
                              void* d_out, int out_size, void* d_ws, size_t ws_size,
                              hipStream_t stream) {
    const float* x         = (const float*)d_in[0];
    const float* gate_w    = (const float*)d_in[1];
    const float* gate_bank = (const float*)d_in[2];
    const float* up_bank   = (const float*)d_in[3];
    const float* down_bank = (const float*)d_in[4];
    float* out = (float*)d_out;

    char* ws = (char*)d_ws;
    int*    counts = (int*)   (ws + WS_COUNT);
    int*    offs   = (int*)   (ws + WS_OFFS);
    int*    list   = (int*)   (ws + WS_LIST);
    int*    sel_e  = (int*)   (ws + WS_SEL_E);
    int*    sel_p  = (int*)   (ws + WS_SEL_P);
    float*  sel_w  = (float*) (ws + WS_SEL_W);
    __bf16* wg_t   = (__bf16*)(ws + WS_WG_T);
    __bf16* wu_t   = (__bf16*)(ws + WS_WU_T);
    __bf16* wd_t   = (__bf16*)(ws + WS_WD_T);
    __bf16* hbuf   = (__bf16*)(ws + WS_H);
    float*  ybuf   = (float*) (ws + WS_Y);

    moe_zero<<<1, 32, 0, stream>>>(counts);
    moe_gate<<<NTOK / 256, 256, 0, stream>>>(x, gate_w, counts, list,
                                             sel_e, sel_p, sel_w);
    moe_offsets<<<1, 1, 0, stream>>>(counts, offs);

    int tgrid = NEXP * (D_IN / 32) * (H_DIM / 32);     // 9216 tiles
    transpose_cvt<<<tgrid, 256, 0, stream>>>(gate_bank, wg_t, D_IN, H_DIM);
    transpose_cvt<<<tgrid, 256, 0, stream>>>(up_bank,   wu_t, D_IN, H_DIM);
    transpose_cvt<<<tgrid, 256, 0, stream>>>(down_bank, wd_t, H_DIM, D_IN);

    moe_gateup<<<NEXP * TILES_PER_E, 256, 0, stream>>>(
        x, wg_t, wu_t, counts, offs, list, hbuf);
    moe_down<<<NEXP * TILES_PER_E, 256, 0, stream>>>(
        hbuf, wd_t, counts, offs, ybuf);
    moe_combine<<<(NTOK * D_IN + 255) / 256, 256, 0, stream>>>(
        ybuf, offs, sel_e, sel_p, sel_w, out);
}
